// LSTM_36429912604934
// MI455X (gfx1250) — compile-verified, matrix operand order checked by reading, not run on hardware
//
#include <hip/hip_runtime.h>

// ---------------------------------------------------------------------------
// Types for WMMA fragments
// ---------------------------------------------------------------------------
typedef __attribute__((ext_vector_type(16))) _Float16 v16h;
typedef __attribute__((ext_vector_type(8)))  _Float16 v8h;
typedef __attribute__((ext_vector_type(8)))  float    v8f;

static constexpr int B  = 64;
static constexpr int T  = 512;
static constexpr int I  = 256;
static constexpr int H  = 1024;
static constexpr int G4 = 4 * H;        // 4096
static constexpr int NBLOCKS  = H / 16; // 64 persistent workgroups
static constexpr int NTHREADS = 256;    // 8 waves: (gate 0..3) x (K-part 0..1)

// ---------------------------------------------------------------------------
// Fragment loader.
//
// 16-bit A (16x32) / B (32x16) VGPR layout (CDNA5 ISA 7.12.2):
//   lanes 0-15  hold row/col (lane&15), elements K = {0..7, 16..23}
//   lanes 16-31 hold row/col (lane&15), elements K = {8..15, 24..31}
// -> per lane: two 16-byte (8 x f16) loads at koff and koff+16,
//    koff = (lane>>4)*8.  v16h element order matches VGPR order.
// ---------------------------------------------------------------------------
__device__ __forceinline__ v16h load_frag_f16(const _Float16* __restrict__ base, int koff) {
    v16h r;
    *reinterpret_cast<v8h*>(&r)       = *reinterpret_cast<const v8h*>(base + koff);
    *(reinterpret_cast<v8h*>(&r) + 1) = *reinterpret_cast<const v8h*>(base + koff + 16);
    return r;
}

__device__ __forceinline__ float sigf(float x) {
    return 1.0f / (1.0f + __expf(-x));
}

// ---------------------------------------------------------------------------
// Grid-wide barrier (monotonic counter, agent scope).  Safe: all 64 WGs are
// co-resident.  Spin uses s_sleep to be polite to the scheduler.
// ---------------------------------------------------------------------------
__device__ __forceinline__ void grid_sync(unsigned* cnt, unsigned& epoch, unsigned nb) {
    __syncthreads();
    if (threadIdx.x == 0) {
        epoch += 1;
        __threadfence();   // release: make this WG's h/c writes visible
        __hip_atomic_fetch_add(cnt, 1u, __ATOMIC_RELEASE, __HIP_MEMORY_SCOPE_AGENT);
        while (__hip_atomic_load(cnt, __ATOMIC_ACQUIRE, __HIP_MEMORY_SCOPE_AGENT) <
               epoch * nb) {
            __builtin_amdgcn_s_sleep(2);
        }
        __threadfence();   // acquire: invalidate L0/L1 so fresh h is seen
    }
    __syncthreads();
}

// ---------------------------------------------------------------------------
// One LSTM step for this WG's 16-column slice of H.
//
// 8 waves: wave = (gate g, kpart p).  p==0 computes x_t @ Wih^T (K=Kx),
// p==1 computes h_{t-1} @ Whh^T (K=H).  Partials meet in LDS, then the
// fused sigmoid/tanh cell update runs and emits h_t as f16 (next step's
// WMMA A-matrix / next layer's input).
// ---------------------------------------------------------------------------
__device__ __forceinline__ void lstm_step_slice(
    const _Float16* __restrict__ xin,    // [B rows, xstride] f16 (base at t)
    long            xstride,
    int             Kx,                  // 256 (layer0) or 1024 (layer1)
    const _Float16* __restrict__ Wih,    // [4H, Kx] f16 row-major
    const _Float16* __restrict__ Whh,    // [4H, H]  f16 row-major
    const _Float16* __restrict__ hprev,  // [B, H] f16
    const float*    __restrict__ bias,   // [4H] combined b_ih+b_hh
    float*          __restrict__ cstate, // [B, H] f32 (slice-private RMW)
    _Float16*       __restrict__ hout,   // [B, H] f16
    float           gsm[4][2][B][16],
    int lane, int gate, int kpart, int col, int koff, int nblk)
{
    v8f acc[4] = {};                     // 4 M-tiles (full batch 64)

    if (kpart == 0) {
        const _Float16* Wb = Wih + (size_t)(gate * H + nblk * 16 + col) * Kx;
        for (int kc = 0; kc < Kx; kc += 32) {
            v16h bfr = load_frag_f16(Wb + kc, koff);
#pragma unroll
            for (int m = 0; m < 4; ++m) {
                v16h afr = load_frag_f16(xin + (long)(m * 16 + col) * xstride + kc, koff);
                acc[m] = __builtin_amdgcn_wmma_f32_16x16x32_f16(
                    false, afr, false, bfr, (short)0, acc[m], false, false);
            }
        }
    } else {
        const _Float16* Wb = Whh + (size_t)(gate * H + nblk * 16 + col) * H;
        for (int kc = 0; kc < H; kc += 32) {
            v16h bfr = load_frag_f16(Wb + kc, koff);
#pragma unroll
            for (int m = 0; m < 4; ++m) {
                v16h afr = load_frag_f16(hprev + (long)(m * 16 + col) * H + kc, koff);
                acc[m] = __builtin_amdgcn_wmma_f32_16x16x32_f16(
                    false, afr, false, bfr, (short)0, acc[m], false, false);
            }
        }
    }

    // Spill accumulators (C/D layout: vgpr r -> M = r (lanes 0-15) or 8+r).
#pragma unroll
    for (int m = 0; m < 4; ++m) {
        const int rbase = m * 16 + ((lane >> 4) << 3);
#pragma unroll
        for (int r = 0; r < 8; ++r)
            gsm[gate][kpart][rbase + r][col] = acc[m][r];
    }
    __syncthreads();

    // Fused gate nonlinearities + cell/hidden update (1024 elems / 256 thr).
    for (int e = threadIdx.x; e < B * 16; e += NTHREADS) {
        const int b  = e >> 4;
        const int n  = e & 15;
        const int gc = nblk * 16 + n;
        const float ig = sigf (gsm[0][0][b][n] + gsm[0][1][b][n] + bias[gc]);
        const float fg = sigf (gsm[1][0][b][n] + gsm[1][1][b][n] + bias[H + gc]);
        const float gg = tanhf(gsm[2][0][b][n] + gsm[2][1][b][n] + bias[2 * H + gc]);
        const float og = sigf (gsm[3][0][b][n] + gsm[3][1][b][n] + bias[3 * H + gc]);
        const float c  = fg * cstate[b * H + gc] + ig * gg;
        cstate[b * H + gc] = c;
        hout[b * H + gc]   = (_Float16)(og * tanhf(c));
    }
    // gsm reuse is protected by the grid_sync's leading __syncthreads().
}

// ---------------------------------------------------------------------------
// Persistent kernel: one launch for the whole T=512, 2-layer recurrence.
// Weights (26 MiB f16) stay L2-resident for the entire loop.
// ---------------------------------------------------------------------------
__global__ __launch_bounds__(NTHREADS) void lstm_persistent_kernel(
    const _Float16* __restrict__ x16,   // [B, T, I] f16
    const _Float16* __restrict__ Wih0, const _Float16* __restrict__ Whh0,
    const _Float16* __restrict__ Wih1, const _Float16* __restrict__ Whh1,
    const float*    __restrict__ bias0, const float* __restrict__ bias1,
    _Float16* __restrict__ h0a, _Float16* __restrict__ h0b,   // ping-pong
    _Float16* __restrict__ h1a, _Float16* __restrict__ h1b,
    float* __restrict__ c0, float* __restrict__ c1,
    unsigned* __restrict__ barcnt)
{
    __shared__ float gsm[4][2][B][16];  // 32 KB

    const int lane  = threadIdx.x & 31;
    const int wv    = threadIdx.x >> 5; // 0..7
    const int gate  = wv >> 1;          // 0..3
    const int kpart = wv & 1;           // 0: input proj, 1: recurrent proj
    const int col   = lane & 15;
    const int koff  = (lane >> 4) << 3; // 0 or 8
    const int nblk  = blockIdx.x;       // H-slice 0..63
    unsigned  epoch = 0;

    for (int t = 0; t < T; ++t) {
        const _Float16* h0s = (t & 1) ? h0b : h0a;
        _Float16*       h0d = (t & 1) ? h0a : h0b;
        const _Float16* h1s = (t & 1) ? h1b : h1a;
        _Float16*       h1d = (t & 1) ? h1a : h1b;

        // layer 0: x_t rows have stride T*I
        lstm_step_slice(x16 + (long)t * I, (long)T * I, I,
                        Wih0, Whh0, h0s, bias0, c0, h0d, gsm,
                        lane, gate, kpart, col, koff, nblk);
        grid_sync(barcnt, epoch, gridDim.x);   // h0d visible to all WGs

        // layer 1: input is layer0's fresh h_t (stride H)
        lstm_step_slice(h0d, (long)H, H,
                        Wih1, Whh1, h1s, bias1, c1, h1d, gsm,
                        lane, gate, kpart, col, koff, nblk);
        grid_sync(barcnt, epoch, gridDim.x);   // h1d visible to all WGs
    }
    // T even -> final hidden states end up in h0a / h1a.
}

// ---------------------------------------------------------------------------
// Helpers: f32->f16 conversion, bias combine, final FC.
// ---------------------------------------------------------------------------
__global__ void cvt_f32_to_f16_kernel(const float* __restrict__ in,
                                      _Float16* __restrict__ out, int n) {
    int i = blockIdx.x * blockDim.x + threadIdx.x;
    if (i < n) out[i] = (_Float16)in[i];
}

__global__ void bias_combine_kernel(const float* __restrict__ a,
                                    const float* __restrict__ b,
                                    float* __restrict__ out, int n) {
    int i = blockIdx.x * blockDim.x + threadIdx.x;
    if (i < n) out[i] = a[i] + b[i];
}

// out[row, c] = sigmoid(relu(hn[row,:]) . fc_w[c,:] + fc_b[c]);
// hn = concat(h0_T, h1_T) -> 128 rows x 2 cols: 256 tiny dot products.
__global__ void final_fc_kernel(const _Float16* __restrict__ h0T,
                                const _Float16* __restrict__ h1T,
                                const float* __restrict__ fc_w,
                                const float* __restrict__ fc_b,
                                float* __restrict__ out) {
    int i = blockIdx.x * blockDim.x + threadIdx.x;
    if (i >= 2 * B * 2) return;
    const int row  = i >> 1;
    const int cdim = i & 1;
    const _Float16* h = (row < B) ? (h0T + (long)row * H)
                                  : (h1T + (long)(row - B) * H);
    float s = fc_b[cdim];
    for (int k = 0; k < H; ++k) {
        float v = (float)h[k];
        v = v > 0.0f ? v : 0.0f;
        s += v * fc_w[cdim * H + k];
    }
    out[row * 2 + cdim] = 1.0f / (1.0f + __expf(-s));
}

// ---------------------------------------------------------------------------
// Host launcher
// ---------------------------------------------------------------------------
static inline size_t align256(size_t x) { return (x + 255) & ~size_t(255); }

extern "C" void kernel_launch(void* const* d_in, const int* in_sizes, int n_in,
                              void* d_out, int out_size, void* d_ws, size_t ws_size,
                              hipStream_t stream) {
    (void)in_sizes; (void)n_in; (void)out_size; (void)ws_size;

    const float* x    = (const float*)d_in[0];
    const float* Wih0 = (const float*)d_in[1];
    const float* Whh0 = (const float*)d_in[2];
    const float* bih0 = (const float*)d_in[3];
    const float* bhh0 = (const float*)d_in[4];
    const float* Wih1 = (const float*)d_in[5];
    const float* Whh1 = (const float*)d_in[6];
    const float* bih1 = (const float*)d_in[7];
    const float* bhh1 = (const float*)d_in[8];
    const float* fc_w = (const float*)d_in[9];
    const float* fc_b = (const float*)d_in[10];
    float* out = (float*)d_out;

    // ---- carve workspace ------------------------------------------------
    char* ws = (char*)d_ws;
    size_t off = 0;
    auto carve = [&](size_t bytes) { void* p = ws + off; off += align256(bytes); return p; };

    _Float16* x16   = (_Float16*)carve((size_t)B * T * I * 2);   // 16 MiB
    _Float16* Wih0h = (_Float16*)carve((size_t)G4 * I * 2);
    _Float16* Whh0h = (_Float16*)carve((size_t)G4 * H * 2);
    _Float16* Wih1h = (_Float16*)carve((size_t)G4 * H * 2);
    _Float16* Whh1h = (_Float16*)carve((size_t)G4 * H * 2);
    float*    bias0 = (float*)   carve((size_t)G4 * 4);
    float*    bias1 = (float*)   carve((size_t)G4 * 4);
    _Float16* h0a   = (_Float16*)carve((size_t)B * H * 2);
    _Float16* h0b   = (_Float16*)carve((size_t)B * H * 2);
    _Float16* h1a   = (_Float16*)carve((size_t)B * H * 2);
    _Float16* h1b   = (_Float16*)carve((size_t)B * H * 2);
    float*    c0    = (float*)   carve((size_t)B * H * 4);
    float*    c1    = (float*)   carve((size_t)B * H * 4);
    unsigned* barcnt= (unsigned*)carve(256);

    // ---- setup ----------------------------------------------------------
    auto cvt = [&](const float* src, _Float16* dst, long n) {
        cvt_f32_to_f16_kernel<<<(unsigned)((n + 255) / 256), 256, 0, stream>>>(src, dst, (int)n);
    };
    cvt(x,    x16,   (long)B * T * I);
    cvt(Wih0, Wih0h, (long)G4 * I);
    cvt(Whh0, Whh0h, (long)G4 * H);
    cvt(Wih1, Wih1h, (long)G4 * H);
    cvt(Whh1, Whh1h, (long)G4 * H);
    bias_combine_kernel<<<(G4 + 255) / 256, 256, 0, stream>>>(bih0, bhh0, bias0, G4);
    bias_combine_kernel<<<(G4 + 255) / 256, 256, 0, stream>>>(bih1, bhh1, bias1, G4);

    hipMemsetAsync(h0a,    0, (size_t)B * H * 2, stream);   // h_{-1} = 0
    hipMemsetAsync(h1a,    0, (size_t)B * H * 2, stream);
    hipMemsetAsync(c0,     0, (size_t)B * H * 4, stream);
    hipMemsetAsync(c1,     0, (size_t)B * H * 4, stream);
    hipMemsetAsync(barcnt, 0, 256, stream);

    // ---- whole recurrence in ONE persistent launch ----------------------
    lstm_persistent_kernel<<<NBLOCKS, NTHREADS, 0, stream>>>(
        x16, Wih0h, Whh0h, Wih1h, Whh1h, bias0, bias1,
        h0a, h0b, h1a, h1b, c0, c1, barcnt);

    // final hidden states are in h0a / h1a (T even)
    final_fc_kernel<<<1, 256, 0, stream>>>(h0a, h1a, fc_w, fc_b, out);
}
